// MultiHeadAttention_86878598463631
// MI455X (gfx1250) — compile-verified, matrix operand order checked by reading, not run on hardware
//
#include <hip/hip_runtime.h>
#include <hip/hip_bf16.h>

// ---------------------------------------------------------------------------
// MHA forward for MI455X (gfx1250, wave32, WMMA).
//   B=2, S=2048, C=1024, H=16, D=64.
//   bf16 V_WMMA_F32_16X16X32_BF16 everywhere, fp32 accumulate.
//   - GEMMs register-blocked (wave = 16x64 tile, 4 WMMAs per A-fragment) AND
//     software-pipelined (double-buffered fragments -> partial s_wait_loadcnt).
//   - Score strip (32 q x 2048 k, fp32) LDS-resident (263KB of 320KB/WGP);
//     attn matrix (536MB) written to HBM exactly once, nontemporal.
//   - attn@V reads exp-scores from LDS; 1/rowsum folded into epilogue.
// ---------------------------------------------------------------------------

#define B_ 2
#define S_ 2048
#define C_ 1024
#define H_ 16
#define D_ 64
#define BH_ (B_ * H_)
#define SROW 2056                      // LDS row stride (f32), 16B aligned
#define ATTN_SMEM (size_t)(32 * SROW * 4 + 32 * 4)

typedef __bf16 bf16_t;
typedef __attribute__((ext_vector_type(4)))  bf16_t v4bf;
typedef __attribute__((ext_vector_type(8)))  bf16_t v8bf;
typedef __attribute__((ext_vector_type(16))) bf16_t v16bf;
typedef __attribute__((ext_vector_type(4)))  float  v4f;
typedef __attribute__((ext_vector_type(8)))  float  v8f;

__device__ __forceinline__ v8f wmma_bf16(v16bf a, v16bf b, v8f c) {
  // D = A(16x32) * B(32x16) + C, f32 accumulate
  return __builtin_amdgcn_wmma_f32_16x16x32_bf16(
      /*neg_a=*/false, a, /*neg_b=*/false, b,
      /*c_mod=*/(short)0, c, /*reuse_a=*/false, /*reuse_b=*/false);
}

// Load a 16x32 bf16 A-fragment (or a K-major B-fragment) per the ISA layout:
// lane L: row m = L&15, kbase = (L&16)?8:0, elements {kb..kb+7, kb+16..kb+23}
// -> two contiguous 16B loads per lane. `stride` = elements between rows.
__device__ __forceinline__ v16bf load_frag(const bf16_t* __restrict__ base,
                                           int stride) {
  const int lane = threadIdx.x & 31;
  const bf16_t* p = base + (size_t)(lane & 15) * stride + ((lane & 16) ? 8 : 0);
  v8bf lo = *(const v8bf*)(p);
  v8bf hi = *(const v8bf*)(p + 16);
  v16bf r;
#pragma unroll
  for (int i = 0; i < 8; ++i) { r[i] = lo[i]; r[8 + i] = hi[i]; }
  return r;
}

// Build a 16x32 A-fragment from fp32 scores in LDS (exp-values), cvt to bf16.
__device__ __forceinline__ v16bf frag_from_lds(const float* __restrict__ base,
                                               int k0) {
  const int lane = threadIdx.x & 31;
  const int kb = (lane & 16) ? 8 : 0;
  const float* p = base + (size_t)(lane & 15) * SROW + k0 + kb;
  v4f x0 = *(const v4f*)(p);
  v4f x1 = *(const v4f*)(p + 4);
  v4f x2 = *(const v4f*)(p + 16);
  v4f x3 = *(const v4f*)(p + 20);
  v16bf r;
#pragma unroll
  for (int i = 0; i < 4; ++i) {
    r[i]      = (bf16_t)x0[i];
    r[4 + i]  = (bf16_t)x1[i];
    r[8 + i]  = (bf16_t)x2[i];
    r[12 + i] = (bf16_t)x3[i];
  }
  return r;
}

// --------------------------- prep kernels ----------------------------------

__global__ void cvt_f32_to_bf16(const float* __restrict__ x,
                                bf16_t* __restrict__ y, int n) {
  int i = (blockIdx.x * blockDim.x + threadIdx.x) * 4;
  if (i >= n) return;
  v4f v = *(const v4f*)(x + i);
  v4bf o;
#pragma unroll
  for (int j = 0; j < 4; ++j) o[j] = (bf16_t)v[j];
  *(v4bf*)(y + i) = o;
}

// wt[n*C + k] = (bf16) w[k*C + n]  (so B-fragments load contiguously in K)
__global__ void transpose_w_bf16(const float* __restrict__ w,
                                 bf16_t* __restrict__ wt) {
  int idx = blockIdx.x * blockDim.x + threadIdx.x;
  if (idx >= C_ * C_) return;
  int n = idx / C_, k = idx % C_;
  wt[idx] = (bf16_t)w[(size_t)k * C_ + n];
}

// ---- shared GEMM core: 16x64 tile per wave, double-buffered fragments -----
// Computes acc[0..3] for the four 16x16 sub-tiles along N.
__device__ __forceinline__ void gemm_core_16x64(
    const bf16_t* __restrict__ xr, const bf16_t* __restrict__ wr, v8f acc[4]) {
  v16bf a  = load_frag(xr, C_);
  v16bf b0 = load_frag(wr, C_);
  v16bf b1 = load_frag(wr + (size_t)16 * C_, C_);
  v16bf b2 = load_frag(wr + (size_t)32 * C_, C_);
  v16bf b3 = load_frag(wr + (size_t)48 * C_, C_);
  for (int k0 = 32; k0 < C_; k0 += 32) {
    // issue next step's 10 b128 loads before consuming current fragments
    __builtin_prefetch(xr + k0 + 96, 0, 3);    // WGP-scope global_prefetch_b8
    v16bf an  = load_frag(xr + k0, C_);
    v16bf bn0 = load_frag(wr + k0, C_);
    v16bf bn1 = load_frag(wr + (size_t)16 * C_ + k0, C_);
    v16bf bn2 = load_frag(wr + (size_t)32 * C_ + k0, C_);
    v16bf bn3 = load_frag(wr + (size_t)48 * C_ + k0, C_);
    acc[0] = wmma_bf16(a, b0, acc[0]);
    acc[1] = wmma_bf16(a, b1, acc[1]);
    acc[2] = wmma_bf16(a, b2, acc[2]);
    acc[3] = wmma_bf16(a, b3, acc[3]);
    a = an; b0 = bn0; b1 = bn1; b2 = bn2; b3 = bn3;
  }
  acc[0] = wmma_bf16(a, b0, acc[0]);
  acc[1] = wmma_bf16(a, b1, acc[1]);
  acc[2] = wmma_bf16(a, b2, acc[2]);
  acc[3] = wmma_bf16(a, b3, acc[3]);
}

// --------------------------- projection GEMM -------------------------------
// out[M=4096, N=1024] = X @ W^T + bias.
// vmode=0: row-major bf16 [B*S, C].  vmode=1: V stored as [B, H, D, S] bf16.
__global__ __launch_bounds__(256) void proj_gemm_bf16(
    const bf16_t* __restrict__ X, const bf16_t* __restrict__ WT,
    const float* __restrict__ bias, bf16_t* __restrict__ out, int vmode) {
  const int w = threadIdx.x >> 5, lane = threadIdx.x & 31;
  const int tm  = blockIdx.y * 8 + w;    // 0..255  (16-row tile)
  const int tn0 = blockIdx.x * 4;        // 0..63   (first of four 16-col tiles)
  const bf16_t* xr = X + (size_t)tm * 16 * C_;
  const bf16_t* wr = WT + (size_t)tn0 * 16 * C_;
  v8f acc[4] = {v8f{}, v8f{}, v8f{}, v8f{}};
  gemm_core_16x64(xr, wr, acc);
  const int n = lane & 15, m0 = (lane & 16) ? 8 : 0;
#pragma unroll
  for (int j = 0; j < 4; ++j) {
    const int ng = (tn0 + j) * 16 + n;
    const float bb = bias[ng];
#pragma unroll
    for (int r = 0; r < 8; ++r) {
      const int mg = tm * 16 + m0 + r;
      const float v = acc[j][r] + bb;
      if (vmode) {
        int bi = mg / S_, s = mg % S_;
        int hh = ng / D_, d = ng % D_;
        out[(((size_t)(bi * H_ + hh)) * D_ + d) * S_ + s] = (bf16_t)v;
      } else {
        out[(size_t)mg * C_ + ng] = (bf16_t)v;
      }
    }
  }
}

// --------------------------- output GEMM (fp32 out) ------------------------
__global__ __launch_bounds__(256) void out_gemm_f32(
    const bf16_t* __restrict__ X, const bf16_t* __restrict__ WT,
    const float* __restrict__ bias, float* __restrict__ out) {
  const int w = threadIdx.x >> 5, lane = threadIdx.x & 31;
  const int tm  = blockIdx.y * 8 + w;
  const int tn0 = blockIdx.x * 4;
  const bf16_t* xr = X + (size_t)tm * 16 * C_;
  const bf16_t* wr = WT + (size_t)tn0 * 16 * C_;
  v8f acc[4] = {v8f{}, v8f{}, v8f{}, v8f{}};
  gemm_core_16x64(xr, wr, acc);
  const int n = lane & 15, m0 = (lane & 16) ? 8 : 0;
#pragma unroll
  for (int j = 0; j < 4; ++j) {
    const int ng = (tn0 + j) * 16 + n;
    const float bb = bias[ng];
#pragma unroll
    for (int r = 0; r < 8; ++r) {
      const int mg = tm * 16 + m0 + r;
      out[(size_t)mg * C_ + ng] = acc[j][r] + bb;
    }
  }
}

// --------------------------- attention kernel ------------------------------
// Grid: (B*H, S/32).  256 threads (8 waves).  LDS: scores[32][SROW] + inv[32].
__global__ __launch_bounds__(256) void attn_kernel(
    const bf16_t* __restrict__ Qh, const bf16_t* __restrict__ Kh,
    const bf16_t* __restrict__ Vt, bf16_t* __restrict__ ctx,
    float* __restrict__ attn_out) {
  extern __shared__ char smem_raw[];
  float* scores  = (float*)smem_raw;          // [32][SROW], exp-values later
  float* row_inv = scores + 32 * SROW;        // [32] 1/rowsum

  const int bh = blockIdx.x;                  // b*H + h
  const int b  = bh / H_;
  const int h  = bh % H_;
  const int q0 = blockIdx.y * 32;
  const int w    = threadIdx.x >> 5;
  const int lane = threadIdx.x & 31;
  const float scale = 0.125f;                 // 1/sqrt(64)

  // ---- scores = scale * Q @ K^T into LDS ----------------------------------
  // Wave holds Q-fragments for BOTH 16-query row tiles; one K-fragment pair
  // feeds 4 WMMAs.  K-fragments double-buffered one 8-stride step ahead.
  {
    const bf16_t* qbase = Qh + ((size_t)(b * S_ + q0)) * C_ + h * D_;
    v16bf aq00 = load_frag(qbase, C_);                         // rt0, d 0..31
    v16bf aq01 = load_frag(qbase + 32, C_);                    // rt0, d 32..63
    v16bf aq10 = load_frag(qbase + (size_t)16 * C_, C_);       // rt1, d 0..31
    v16bf aq11 = load_frag(qbase + (size_t)16 * C_ + 32, C_);  // rt1, d 32..63
    const int n = lane & 15, m0 = (lane & 16) ? 8 : 0;
    const bf16_t* khead = Kh + ((size_t)(b * S_)) * C_ + h * D_;
    constexpr int NCT = (S_ / 16) / 8;        // 16 iterations per wave
    int ct = w;
    v16bf bk0 = load_frag(khead + (size_t)ct * 16 * C_, C_);
    v16bf bk1 = load_frag(khead + (size_t)ct * 16 * C_ + 32, C_);
    for (int it = 0; it < NCT; ++it) {
      const int ctn = ct + 8;
      v16bf nk0 = bk0, nk1 = bk1;
      if (it + 1 < NCT) {                     // prefetch next K tile
        nk0 = load_frag(khead + (size_t)ctn * 16 * C_, C_);
        nk1 = load_frag(khead + (size_t)ctn * 16 * C_ + 32, C_);
      }
      v8f acc0 = {}, acc1 = {};
      acc0 = wmma_bf16(aq00, bk0, acc0);
      acc1 = wmma_bf16(aq10, bk0, acc1);
      acc0 = wmma_bf16(aq01, bk1, acc0);
      acc1 = wmma_bf16(aq11, bk1, acc1);
#pragma unroll
      for (int r = 0; r < 8; ++r) {
        scores[(size_t)(m0 + r) * SROW + ct * 16 + n]      = acc0[r] * scale;
        scores[(size_t)(16 + m0 + r) * SROW + ct * 16 + n] = acc1[r] * scale;
      }
      bk0 = nk0; bk1 = nk1; ct = ctn;
    }
  }
  __syncthreads();

  // ---- softmax (rows 4w..4w+3 per wave); write normalized attn once -------
  for (int rr = 0; rr < 4; ++rr) {
    const int row = w * 4 + rr;
    float* srow = scores + (size_t)row * SROW;
    float mx = -3.4e38f;
    for (int i = lane; i < S_; i += 32) mx = fmaxf(mx, srow[i]);
#pragma unroll
    for (int off = 16; off > 0; off >>= 1)
      mx = fmaxf(mx, __shfl_xor(mx, off, 32));
    float sum = 0.f;
    for (int i = lane; i < S_; i += 32) {
      float e = __expf(srow[i] - mx);
      srow[i] = e;                            // keep unnormalized exp in LDS
      sum += e;
    }
#pragma unroll
    for (int off = 16; off > 0; off >>= 1) sum += __shfl_xor(sum, off, 32);
    const float inv = 1.f / sum;
    if (lane == 0) row_inv[row] = inv;
    float* dst = attn_out + ((size_t)bh * S_ + q0 + row) * S_;
    for (int i = lane; i < S_; i += 32)
      __builtin_nontemporal_store(srow[i] * inv, dst + i);  // streamed, once
  }
  __syncthreads();

  // ---- ctx = (exp_scores @ V) * inv  (wave w -> row tile w>>2, d tile w&3) -
  // V fragment (global) and score fragment (LDS) double-buffered.
  {
    const int rt = w >> 2, dt = w & 3;
    const float*  sbase = scores + (size_t)rt * 16 * SROW;
    const bf16_t* vbase = Vt + ((size_t)bh * D_ + dt * 16) * S_;
    v8f acc = {};
    v16bf a  = frag_from_lds(sbase, 0);
    v16bf bv = load_frag(vbase, S_);
    for (int k0 = 32; k0 < S_; k0 += 32) {
      v16bf an = frag_from_lds(sbase, k0);
      v16bf bn = load_frag(vbase + k0, S_);
      acc = wmma_bf16(a, bv, acc);
      a = an; bv = bn;
    }
    acc = wmma_bf16(a, bv, acc);
    const int n = lane & 15, m0 = (lane & 16) ? 8 : 0;
#pragma unroll
    for (int r = 0; r < 8; ++r) {
      const int row = rt * 16 + m0 + r;
      const float v = acc[r] * row_inv[row];
      const int sg = q0 + row;
      ctx[((size_t)(b * S_ + sg)) * C_ + h * D_ + dt * 16 + n] = (bf16_t)v;
    }
  }
}

// --------------------------- host launcher ---------------------------------
extern "C" void kernel_launch(void* const* d_in, const int* in_sizes, int n_in,
                              void* d_out, int out_size, void* d_ws,
                              size_t ws_size, hipStream_t stream) {
  const float* q  = (const float*)d_in[0];
  const float* k  = (const float*)d_in[1];
  const float* v  = (const float*)d_in[2];
  const float* wq = (const float*)d_in[3];
  const float* bq = (const float*)d_in[4];
  const float* wk = (const float*)d_in[5];
  const float* bk = (const float*)d_in[6];
  const float* wv = (const float*)d_in[7];
  const float* bv = (const float*)d_in[8];
  const float* wo = (const float*)d_in[9];
  const float* bo = (const float*)d_in[10];

  float* out_f32  = (float*)d_out;                              // [B,S,C]
  float* attn_f32 = out_f32 + (size_t)B_ * S_ * C_;             // [B,H,S,S]

  // Workspace layout (bf16): 4 transposed weights + q/k/v + Qh/Kh + Vt + ctx
  constexpr size_t SZ_W = (size_t)C_ * C_;        // 1M elems
  constexpr size_t SZ_X = (size_t)B_ * S_ * C_;   // 4M elems
  bf16_t* p    = (bf16_t*)d_ws;
  bf16_t* wqT  = p; p += SZ_W;
  bf16_t* wkT  = p; p += SZ_W;
  bf16_t* wvT  = p; p += SZ_W;
  bf16_t* woT  = p; p += SZ_W;
  bf16_t* qb   = p; p += SZ_X;
  bf16_t* kb   = p; p += SZ_X;
  bf16_t* vb   = p; p += SZ_X;
  bf16_t* Qh   = p; p += SZ_X;
  bf16_t* Kh   = p; p += SZ_X;
  bf16_t* Vt   = p; p += SZ_X;   // [B,H,D,S]
  bf16_t* ctx  = p; p += SZ_X;   // total 64 MB

  // 1) weights -> bf16, transposed
  {
    int n = C_ * C_;
    dim3 g((n + 255) / 256), blk(256);
    transpose_w_bf16<<<g, blk, 0, stream>>>(wq, wqT);
    transpose_w_bf16<<<g, blk, 0, stream>>>(wk, wkT);
    transpose_w_bf16<<<g, blk, 0, stream>>>(wv, wvT);
    transpose_w_bf16<<<g, blk, 0, stream>>>(wo, woT);
  }
  // 2) inputs -> bf16
  {
    int n = (int)SZ_X;
    dim3 g((n / 4 + 255) / 256), blk(256);
    cvt_f32_to_bf16<<<g, blk, 0, stream>>>(q, qb, n);
    cvt_f32_to_bf16<<<g, blk, 0, stream>>>(k, kb, n);
    cvt_f32_to_bf16<<<g, blk, 0, stream>>>(v, vb, n);
  }
  // 3) projections (M=4096 tokens, N=K=1024; waves do 16x64 tiles)
  {
    dim3 g(C_ / 64, (B_ * S_) / 128), blk(256);
    proj_gemm_bf16<<<g, blk, 0, stream>>>(qb, wqT, bq, Qh, 0);
    proj_gemm_bf16<<<g, blk, 0, stream>>>(kb, wkT, bk, Kh, 0);
    proj_gemm_bf16<<<g, blk, 0, stream>>>(vb, wvT, bv, Vt, 1);
  }
  // 4) attention (scores strip LDS-resident; 263KB dynamic LDS)
  {
    hipFuncSetAttribute((const void*)attn_kernel,
                        hipFuncAttributeMaxDynamicSharedMemorySize,
                        (int)ATTN_SMEM);
    dim3 g(BH_, S_ / 32), blk(256);
    attn_kernel<<<g, blk, ATTN_SMEM, stream>>>(Qh, Kh, Vt, ctx, attn_f32);
  }
  // 5) output projection (fp32 out + bias)
  {
    dim3 g(C_ / 64, (B_ * S_) / 128), blk(256);
    out_gemm_f32<<<g, blk, 0, stream>>>(ctx, woT, bo, out_f32);
  }
  (void)in_sizes; (void)n_in; (void)out_size; (void)ws_size;
}
